// SwitchFFN_82471962018205
// MI455X (gfx1250) — matrix operand order repaired and checked by reading, not compiled, not run on hardware
//
#include <hip/hip_runtime.h>
#include <math.h>

#define NTOK 8192
#define DIMD 1024
#define NEXP 8
#define DFFD 4096
#define CAPC 1280
#define NBLK 32            // NTOK / 256
#define LDA  40            // LDS row stride in bf16 elems (80B, 16B-aligned)

typedef __attribute__((ext_vector_type(16))) __bf16 v16bf;
typedef __attribute__((ext_vector_type(8)))  float  v8f;

union FragBF {
    v16bf v;
    uint4 q[2];
};

__device__ __forceinline__ unsigned short f2bf(float f) {
    unsigned u = __builtin_bit_cast(unsigned, f);
    unsigned r = u + 0x7FFFu + ((u >> 16) & 1u);   // round-to-nearest-even
    return (unsigned short)(r >> 16);
}

// branch-free tanh-GELU: e = exp(-2|u|) in (0,1], t = (1-e)/(1+e)
__device__ __forceinline__ float gelu_tanh(float x) {
    float u = 0.7978845608028654f * (x + 0.044715f * x * x * x);
    float a = fabsf(u);
    float e = __expf(-2.0f * a);
    float t = (1.0f - e) * __builtin_amdgcn_rcpf(1.0f + e);
    t = copysignf(t, u);
    return 0.5f * x * (1.0f + t);
}

// ---------------------------------------------------------------- init
__global__ void init_kernel(int* slot2tok, float* probs_colsum) {
    int i = blockIdx.x * 256 + threadIdx.x;
    if (i < NEXP * CAPC) slot2tok[i] = NTOK;
    if (i < NEXP) probs_colsum[i] = 0.0f;
}

// ---------------------------------------------------------------- router
__global__ __launch_bounds__(256) void router_kernel(
    const float* __restrict__ x, const float* __restrict__ Wr,
    float* __restrict__ gate, int* __restrict__ eidx,
    float* __restrict__ probs_colsum)
{
    const int t   = blockIdx.x;
    const int tid = threadIdx.x;
    const float* xrow = x + (size_t)t * DIMD;

    float acc[NEXP];
#pragma unroll
    for (int e = 0; e < NEXP; ++e) acc[e] = 0.0f;

    for (int d = tid; d < DIMD; d += 256) {
        float xv = xrow[d];
        const float* wrow = Wr + d * NEXP;
#pragma unroll
        for (int e = 0; e < NEXP; ++e) acc[e] += xv * wrow[e];
    }
#pragma unroll
    for (int m = 16; m >= 1; m >>= 1) {
#pragma unroll
        for (int e = 0; e < NEXP; ++e) acc[e] += __shfl_xor(acc[e], m, 32);
    }
    __shared__ float red[8][NEXP];
    const int wave = tid >> 5, lane = tid & 31;
    if (lane == 0) {
#pragma unroll
        for (int e = 0; e < NEXP; ++e) red[wave][e] = acc[e];
    }
    __syncthreads();
    if (tid == 0) {
        float lg[NEXP];
#pragma unroll
        for (int e = 0; e < NEXP; ++e) {
            float s = 0.0f;
            for (int w = 0; w < 8; ++w) s += red[w][e];
            lg[e] = s;
        }
        float mx = lg[0];
#pragma unroll
        for (int e = 1; e < NEXP; ++e) mx = fmaxf(mx, lg[e]);
        float p[NEXP], sum = 0.0f;
#pragma unroll
        for (int e = 0; e < NEXP; ++e) { p[e] = __expf(lg[e] - mx); sum += p[e]; }
        float inv = 1.0f / sum;
        int am = 0; float best = lg[0];
#pragma unroll
        for (int e = 1; e < NEXP; ++e) if (lg[e] > best) { best = lg[e]; am = e; }
        gate[t] = p[am] * inv;
        eidx[t] = am;
#pragma unroll
        for (int e = 0; e < NEXP; ++e) atomicAdd(&probs_colsum[e], p[e] * inv);
    }
}

// ---------------------------------------------------------------- order
__global__ __launch_bounds__(256) void order_kernel(
    const int* __restrict__ eidx, int* __restrict__ local_pos,
    int* __restrict__ blockcnt)
{
    __shared__ int se[256];
    const int tid = threadIdx.x;
    const int t = blockIdx.x * 256 + tid;
    se[tid] = eidx[t];
    __syncthreads();
    int e = se[tid];
    int p = 0;
    for (int j = 0; j < tid; ++j) p += (se[j] == e);
    local_pos[t] = p;
    if (tid < NEXP) {
        int c = 0;
        for (int j = 0; j < 256; ++j) c += (se[j] == tid);
        blockcnt[blockIdx.x * NEXP + tid] = c;
    }
}

// ---------------------------------------------------------------- scan + aux
__global__ void scan_aux_kernel(const int* __restrict__ blockcnt,
                                int* __restrict__ baseoff,
                                const float* __restrict__ probs_colsum,
                                float* __restrict__ out_aux)
{
    __shared__ float contrib[NEXP];
    const int e = threadIdx.x;
    if (e < NEXP) {
        int run = 0;
        for (int b = 0; b < NBLK; ++b) {
            baseoff[b * NEXP + e] = run;
            run += blockcnt[b * NEXP + e];
        }
        int tpe = run < CAPC ? run : CAPC;
        contrib[e] = ((float)tpe / (float)NTOK) * (probs_colsum[e] / (float)NTOK);
    }
    __syncthreads();
    if (e == 0) {
        float s = 0.0f;
        for (int i = 0; i < NEXP; ++i) s += contrib[i];
        *out_aux = 0.01f * (float)NEXP * s;
    }
}

// ---------------------------------------------------------------- assign
__global__ __launch_bounds__(256) void assign_kernel(
    const int* __restrict__ eidx, const int* __restrict__ local_pos,
    const int* __restrict__ baseoff, const float* __restrict__ gate,
    float* __restrict__ gate_masked, int* __restrict__ slot2tok)
{
    const int t = blockIdx.x * 256 + threadIdx.x;
    const int e = eidx[t];
    const int pos = baseoff[blockIdx.x * NEXP + e] + local_pos[t];
    if (pos < CAPC) {
        slot2tok[e * CAPC + pos] = t;
        gate_masked[t] = gate[t];
    } else {
        gate_masked[t] = 0.0f;
    }
}

// ---------------------------------------------------------------- gather x -> bf16 slots
__global__ __launch_bounds__(256) void gather_kernel(
    const float* __restrict__ x, const int* __restrict__ slot2tok,
    unsigned short* __restrict__ gx)
{
    const int slot = blockIdx.x;                 // e*CAPC + p
    const int tok = slot2tok[slot];
    unsigned short* dst = gx + (size_t)slot * DIMD;
    if (tok < NTOK) {
        const float* src = x + (size_t)tok * DIMD;
        for (int d = threadIdx.x; d < DIMD; d += 256) dst[d] = f2bf(src[d]);
    } else {
        for (int d = threadIdx.x; d < DIMD; d += 256) dst[d] = 0;
    }
}

// ---------------------------------------------------------------- passthrough copy
__global__ __launch_bounds__(256) void copy_kernel(const float* __restrict__ src,
                                                   float* __restrict__ dst)
{
    const int n4 = NTOK * DIMD / 4;
    const int stride = gridDim.x * blockDim.x;
    for (int i = blockIdx.x * blockDim.x + threadIdx.x; i < n4; i += stride) {
        reinterpret_cast<float4*>(dst)[i] = reinterpret_cast<const float4*>(src)[i];
    }
}

// -------- fragment loaders: contiguous K order k = 16*half + s, same for A and B (wave32) -----------
// A 16x32 bf16: lanes 0-15 row M=lane, K 0..15; lanes 16-31 K 16..31 (contiguous, matches B)
__device__ __forceinline__ FragBF load_fragA(const unsigned short* rowBase, int half) {
    FragBF f;
    f.q[0] = *reinterpret_cast<const uint4*>(rowBase + (half ? 16 : 0));
    f.q[1] = *reinterpret_cast<const uint4*>(rowBase + (half ? 24 : 8));
    return f;
}
// B 32x16 bf16 (stored transposed [N][K] in LDS): lanes 0-15 col N=lane, K 0..15;
// lanes 16-31 col N=lane-16, K 16..31 (16 contiguous K per lane)
__device__ __forceinline__ FragBF load_fragB(const unsigned short* colBase, int half) {
    FragBF f;
    const unsigned short* p = colBase + (half ? 16 : 0);
    f.q[0] = *reinterpret_cast<const uint4*>(p);
    f.q[1] = *reinterpret_cast<const uint4*>(p + 8);
    return f;
}

// ------- shared GEMM body: 128(M) x 64(N) block tile, waves 2(M) x 4(N), ----
// ------- 4 M-subtiles/wave sharing one B fragment = 4 WMMAs per K-step ------

// stage B tile (32K x 64N, f32 -> bf16, transposed [N][K], paired-K b32 stores)
__device__ __forceinline__ void stageB(const float* __restrict__ We, int ld,
                                       int k0, int n0, int tid,
                                       unsigned short* ldsB, int kmax) {
#pragma unroll
    for (int i = 0; i < 4; ++i) {
        int idx = tid + (i << 8);          // 0..1023 pairs
        int n = idx & 63;
        int k = (idx >> 6) << 1;           // 0,2,..,30
        unsigned lo = f2bf(We[(size_t)(k0 + k) * ld + n0 + n]);
        unsigned hi = f2bf(We[(size_t)(k0 + k + 1) * ld + n0 + n]);
        *reinterpret_cast<unsigned*>(&ldsB[n * LDA + k]) = lo | (hi << 16);
    }
    // prefetch next K-step's weight tile (L2-resident; lowers to global_prefetch_b8)
    if (k0 + 32 < kmax) {
        __builtin_prefetch(&We[(size_t)(k0 + 32 + (tid >> 3)) * ld + n0 + ((tid & 7) << 3)], 0, 0);
    }
}

// ---------------------------------------------------------------- GEMM1: h = gelu(gx @ W1 + b1)
__global__ __launch_bounds__(256) void gemm1_kernel(
    const unsigned short* __restrict__ gx, const float* __restrict__ W1,
    const float* __restrict__ b1, unsigned short* __restrict__ h)
{
    const int e  = blockIdx.z;
    const int n0 = blockIdx.x * 64;
    const int m0 = blockIdx.y * 128;
    const int tid = threadIdx.x;
    const int wave = tid >> 5, lane = tid & 31;
    const int waveM = wave >> 2, waveN = wave & 3;
    const int half = lane >> 4, lrow = lane & 15;

    __shared__ unsigned short ldsA[128 * LDA];
    __shared__ unsigned short ldsB[64 * LDA];

    const float* W1e = W1 + (size_t)e * DIMD * DFFD;
    const unsigned short* gxe = gx + ((size_t)e * CAPC + m0) * DIMD;

    v8f c[4];
#pragma unroll
    for (int j = 0; j < 4; ++j) c[j] = (v8f){0.f,0.f,0.f,0.f,0.f,0.f,0.f,0.f};

    for (int k0 = 0; k0 < DIMD; k0 += 32) {
        // stage A: 128 rows x 32 K bf16, 16 elems/thread as uint2
#pragma unroll
        for (int i = 0; i < 4; ++i) {
            int idx = tid + (i << 8);      // 0..1023
            int r = idx >> 3, kk = (idx & 7) << 2;
            uint2 v = *reinterpret_cast<const uint2*>(gxe + (size_t)r * DIMD + k0 + kk);
            *reinterpret_cast<uint2*>(&ldsA[r * LDA + kk]) = v;
        }
        stageB(W1e, DFFD, k0, n0, tid, ldsB, DIMD);
        __syncthreads();
        FragBF B = load_fragB(&ldsB[(waveN * 16 + lrow) * LDA], half);
#pragma unroll
        for (int j = 0; j < 4; ++j) {
            FragBF A = load_fragA(&ldsA[(waveM * 64 + j * 16 + lrow) * LDA], half);
            c[j] = __builtin_amdgcn_wmma_f32_16x16x32_bf16(false, A.v, false, B.v,
                                                           (short)0, c[j], false, false);
        }
        __syncthreads();
    }
    const int col = n0 + waveN * 16 + lrow;
    const float bv = b1[(size_t)e * DFFD + col];
    unsigned short* he = h + (size_t)e * CAPC * DFFD;
#pragma unroll
    for (int j = 0; j < 4; ++j) {
#pragma unroll
        for (int v = 0; v < 8; ++v) {
            int r = m0 + waveM * 64 + j * 16 + v + (half << 3);
            he[(size_t)r * DFFD + col] = f2bf(gelu_tanh(c[j][v] + bv));
        }
    }
}

// ---------------------------------------------------------------- GEMM2: out[tok] = (h @ W2 + b2) * gate
__global__ __launch_bounds__(256) void gemm2_kernel(
    const unsigned short* __restrict__ h, const float* __restrict__ W2,
    const float* __restrict__ b2, const int* __restrict__ slot2tok,
    const float* __restrict__ gate_masked, float* __restrict__ out)
{
    const int e  = blockIdx.z;
    const int n0 = blockIdx.x * 64;
    const int m0 = blockIdx.y * 128;
    const int tid = threadIdx.x;
    const int wave = tid >> 5, lane = tid & 31;
    const int waveM = wave >> 2, waveN = wave & 3;
    const int half = lane >> 4, lrow = lane & 15;

    __shared__ unsigned short ldsA[128 * LDA];
    __shared__ unsigned short ldsB[64 * LDA];
    __shared__ int ldsTok[128];

    if (tid < 128) ldsTok[tid] = slot2tok[e * CAPC + m0 + tid];

    const float* W2e = W2 + (size_t)e * DFFD * DIMD;
    const unsigned short* he = h + ((size_t)e * CAPC + m0) * DFFD;

    v8f c[4];
#pragma unroll
    for (int j = 0; j < 4; ++j) c[j] = (v8f){0.f,0.f,0.f,0.f,0.f,0.f,0.f,0.f};

    for (int k0 = 0; k0 < DFFD; k0 += 32) {
#pragma unroll
        for (int i = 0; i < 4; ++i) {
            int idx = tid + (i << 8);
            int r = idx >> 3, kk = (idx & 7) << 2;
            uint2 v = *reinterpret_cast<const uint2*>(he + (size_t)r * DFFD + k0 + kk);
            *reinterpret_cast<uint2*>(&ldsA[r * LDA + kk]) = v;
        }
        stageB(W2e, DIMD, k0, n0, tid, ldsB, DFFD);
        __syncthreads();
        FragBF B = load_fragB(&ldsB[(waveN * 16 + lrow) * LDA], half);
#pragma unroll
        for (int j = 0; j < 4; ++j) {
            FragBF A = load_fragA(&ldsA[(waveM * 64 + j * 16 + lrow) * LDA], half);
            c[j] = __builtin_amdgcn_wmma_f32_16x16x32_bf16(false, A.v, false, B.v,
                                                           (short)0, c[j], false, false);
        }
        __syncthreads();
    }
    const int col = n0 + waveN * 16 + lrow;
    const float bv = b2[(size_t)e * DIMD + col];
#pragma unroll
    for (int j = 0; j < 4; ++j) {
#pragma unroll
        for (int v = 0; v < 8; ++v) {
            int lr = waveM * 64 + j * 16 + v + (half << 3);
            int tk = ldsTok[lr];
            if (tk < NTOK) out[(size_t)tk * DIMD + col] = (c[j][v] + bv) * gate_masked[tk];
        }
    }
}

// ---------------------------------------------------------------- launch
extern "C" void kernel_launch(void* const* d_in, const int* in_sizes, int n_in,
                              void* d_out, int out_size, void* d_ws, size_t ws_size,
                              hipStream_t stream) {
    const float* x  = (const float*)d_in[0];
    const float* Wr = (const float*)d_in[1];
    const float* W1 = (const float*)d_in[2];
    const float* b1 = (const float*)d_in[3];
    const float* W2 = (const float*)d_in[4];
    const float* b2 = (const float*)d_in[5];
    float* out = (float*)d_out;

    char* ws = (char*)d_ws;
    size_t off = 0;
    unsigned short* h  = (unsigned short*)(ws + off); off += (size_t)NEXP * CAPC * DFFD * 2; // 83.9 MB
    unsigned short* gx = (unsigned short*)(ws + off); off += (size_t)NEXP * CAPC * DIMD * 2; // 21.0 MB
    int*   slot2tok    = (int*)  (ws + off); off += (size_t)NEXP * CAPC * 4;                  // 40 KB
    float* gate        = (float*)(ws + off); off += (size_t)NTOK * 4;
    float* gate_masked = (float*)(ws + off); off += (size_t)NTOK * 4;
    int*   eidx        = (int*)  (ws + off); off += (size_t)NTOK * 4;
    int*   local_pos   = (int*)  (ws + off); off += (size_t)NTOK * 4;
    int*   blockcnt    = (int*)  (ws + off); off += 1024;
    int*   baseoff     = (int*)  (ws + off); off += 1024;
    float* probs_colsum= (float*)(ws + off); off += 256;

    init_kernel<<<dim3((NEXP * CAPC + 255) / 256), 256, 0, stream>>>(slot2tok, probs_colsum);
    router_kernel<<<dim3(NTOK), 256, 0, stream>>>(x, Wr, gate, eidx, probs_colsum);
    order_kernel<<<dim3(NBLK), 256, 0, stream>>>(eidx, local_pos, blockcnt);
    scan_aux_kernel<<<dim3(1), 32, 0, stream>>>(blockcnt, baseoff, probs_colsum,
                                                out + (size_t)NTOK * DIMD);
    assign_kernel<<<dim3(NBLK), 256, 0, stream>>>(eidx, local_pos, baseoff, gate,
                                                  gate_masked, slot2tok);
    gather_kernel<<<dim3(NEXP * CAPC), 256, 0, stream>>>(x, slot2tok, gx);
    copy_kernel<<<dim3(2048), 256, 0, stream>>>(x, out);
    gemm1_kernel<<<dim3(DFFD / 64, CAPC / 128, NEXP), 256, 0, stream>>>(gx, W1, b1, h);
    gemm2_kernel<<<dim3(DIMD / 64, CAPC / 128, NEXP), 256, 0, stream>>>(h, W2, b2, slot2tok,
                                                                        gate_masked, out);
}
